// TransformerEncoderLayer_4595615007197
// MI455X (gfx1250) — matrix operand order repaired and checked by reading, not compiled, not run on hardware
//
#include <hip/hip_runtime.h>
#include <hip/hip_bf16.h>
#include <math.h>

// ---------------------------------------------------------------------------
// Transformer encoder layer forward for MI455X (gfx1250, wave32, WMMA).
// All GEMMs run through v_wmma_f32_16x16x32_bf16 with fp32 accumulation.
// fp32 -> bf16 conversion happens during LDS staging (v_cvt_pk_bf16_f32);
// global loads of the next K-chunk are software-pipelined against WMMA.
// ---------------------------------------------------------------------------

typedef __bf16 bf16_t;
typedef bf16_t v16bf __attribute__((ext_vector_type(16)));
typedef bf16_t v8bf  __attribute__((ext_vector_type(8)));
typedef bf16_t v2bf  __attribute__((ext_vector_type(2)));
typedef float  v8f   __attribute__((ext_vector_type(8)));
typedef float  v2f   __attribute__((ext_vector_type(2)));

#define Bsz 8
#define Nsq 1024
#define Cch 768
#define Hh  12
#define HDd 64
#define FFd 2048

enum { EPI_NONE = 0, EPI_BIAS = 1, EPI_BIAS_GELU = 2, EPI_BIAS_RES = 3, EPI_SCALE = 4 };

// Packed f32x2 -> bf16x2 : single v_cvt_pk_bf16_f32
__device__ __forceinline__ unsigned int pack2(float x, float y) {
    v2f f; f.x = x; f.y = y;
    v2bf b = __builtin_convertvector(f, v2bf);
    return __builtin_bit_cast(unsigned int, b);
}

#define BM 128
#define BN 64
#define BK 32
#define LDA_S 40   // padded LDS row (elements); 80B row = 16B aligned, spreads banks
#define LDB_S 40

// Generic batched GEMM:  C[b,h] = epi( A[b,h] (MxK) * B[b,h] (KxN) )
// A element (m,k) at  A + b*a_off_b + h*a_off_h + m*lda + k
// B element (k,n) at  B + b*b_off_b + h*b_off_h + k*b_ks + n*b_ns
template <int EPI>
__global__ __launch_bounds__(256) void gemm_wmma(
    const float* __restrict__ A, long long a_off_b, long long a_off_h, int lda,
    const float* __restrict__ Bm, long long b_off_b, long long b_off_h, int b_ks, int b_ns,
    float* __restrict__ Cm, long long c_off_b, long long c_off_h, int ldc,
    const float* __restrict__ bias,
    const float* __restrict__ res, int ldres,
    float scale, int K)
{
    __shared__ __align__(16) bf16_t Asm[BM * LDA_S];        // [m][k]
    __shared__ __align__(16) bf16_t Bsm[BN * LDB_S];        // transposed: [n][k]

    const int tid  = threadIdx.x;
    const int wave = tid >> 5;
    const int lane = tid & 31;
    const int bb   = blockIdx.z / Hh;
    const int hh   = blockIdx.z % Hh;

    const float* Ag = A  + (long long)bb * a_off_b + (long long)hh * a_off_h;
    const float* Bg = Bm + (long long)bb * b_off_b + (long long)hh * b_off_h;
    float*       Cg = Cm + (long long)bb * c_off_b + (long long)hh * c_off_h;

    const int row0 = blockIdx.y * BM;
    const int col0 = blockIdx.x * BN;

    // ---- per-thread staging coordinates (fixed for whole kernel) ----
    const int r0   = tid >> 4;              // A: starting row within tile
    const int c0   = (tid << 1) & 31;       // A: even column
    const int kp_t = (tid >> 6) << 1;       // B: starting (even) k within tile
    const int nn_t = tid & 63;              // B: column

    const float* ap = Ag + (long long)(row0 + r0) * lda + c0;          // advance += BK
    const float* bp = Bg + (long long)kp_t * b_ks + (long long)(col0 + nn_t) * b_ns; // += BK*b_ks
    const long long a_i_step = (long long)16 * lda;   // 16 rows per sub-iteration
    const long long b_i_step = (long long)8 * b_ks;   // 8 k (4 pair-bases) per sub-iteration
    const long long b_k_step = (long long)BK * b_ks;
    const long long b_ks_ll  = (long long)b_ks;

    v8f acc[4] = {};                        // 4 tiles of 16x16, fp32 accum

    const int hi   = lane >> 4;             // upper/lower half-wave
    const int arow = (wave << 4) + (lane & 15);
    const int k0a  = hi << 3;               // A frag: halves 0-7 -> k0a.., 8-15 -> +16
    const int k0b  = hi << 3;               // B frag: halves 0-7 -> k0b.., 8-15 -> +16

    float2 areg[8];
    float2 breg[4];
    // ---- prologue: load K-chunk 0 into registers ----
    #pragma unroll
    for (int i = 0; i < 8; ++i) areg[i] = *(const float2*)(ap + i * a_i_step);
    #pragma unroll
    for (int i = 0; i < 4; ++i) {
        breg[i].x = bp[i * b_i_step];
        breg[i].y = bp[i * b_i_step + b_ks_ll];
    }

    const int nk = K / BK;
    for (int kb = 0; kb < nk; ++kb) {
        // ---- commit staged registers to LDS (bf16, packed pairs) ----
        #pragma unroll
        for (int i = 0; i < 8; ++i)
            *(unsigned int*)&Asm[(r0 + 16 * i) * LDA_S + c0] = pack2(areg[i].x, areg[i].y);
        #pragma unroll
        for (int i = 0; i < 4; ++i)
            *(unsigned int*)&Bsm[nn_t * LDB_S + kp_t + 8 * i] = pack2(breg[i].x, breg[i].y);
        __syncthreads();

        // ---- issue next chunk's global loads (overlap with WMMA below) ----
        if (kb + 1 < nk) {
            ap += BK;
            bp += b_k_step;
            #pragma unroll
            for (int i = 0; i < 8; ++i) areg[i] = *(const float2*)(ap + i * a_i_step);
            #pragma unroll
            for (int i = 0; i < 4; ++i) {
                breg[i].x = bp[i * b_i_step];
                breg[i].y = bp[i * b_i_step + b_ks_ll];
            }
        }

        // ---- A fragment (ISA 16-bit A 16x32 layout) ----
        v16bf afrag;
        {
            const bf16_t* app = &Asm[arow * LDA_S];
            v8bf alo = *(const v8bf*)(app + k0a);
            v8bf ahi = *(const v8bf*)(app + k0a + 16);
            #pragma unroll
            for (int i = 0; i < 8; ++i) { afrag[i] = alo[i]; afrag[i + 8] = ahi[i]; }
        }
        // ---- load all 4 B fragments (independent regs), then 4 WMMAs ----
        v16bf bfrag[4];
        #pragma unroll
        for (int t = 0; t < 4; ++t) {
            const bf16_t* bpp = &Bsm[(t * 16 + (lane & 15)) * LDB_S + k0b];
            v8bf blo = *(const v8bf*)bpp;
            v8bf bhi = *(const v8bf*)(bpp + 16);
            #pragma unroll
            for (int i = 0; i < 8; ++i) { bfrag[t][i] = blo[i]; bfrag[t][i + 8] = bhi[i]; }
        }
        #pragma unroll
        for (int t = 0; t < 4; ++t)
            acc[t] = __builtin_amdgcn_wmma_f32_16x16x32_bf16(
                false, afrag, false, bfrag[t], (short)0, acc[t], false, false);
        __syncthreads();
    }

    // ---- epilogue: C/D layout: VGPR r -> M = r + 8*hi; lane&15 -> N ----
    #pragma unroll
    for (int t = 0; t < 4; ++t) {
        const int ncol = col0 + t * 16 + (lane & 15);
        float bv = 0.0f;
        if (EPI == EPI_BIAS || EPI == EPI_BIAS_GELU || EPI == EPI_BIAS_RES) bv = bias[ncol];
        #pragma unroll
        for (int r = 0; r < 8; ++r) {
            const int mrow = row0 + (wave << 4) + (hi << 3) + r;
            float v = acc[t][r];
            if (EPI == EPI_SCALE) v *= scale;
            if (EPI == EPI_BIAS || EPI == EPI_BIAS_GELU || EPI == EPI_BIAS_RES) v += bv;
            if (EPI == EPI_BIAS_GELU) v = 0.5f * v * (1.0f + erff(v * 0.70710678118654752f));
            if (EPI == EPI_BIAS_RES) v += res[(long long)mrow * ldres + ncol];
            Cg[(long long)mrow * ldc + ncol] = v;
        }
    }
}

// LayerNorm over C=768, one block per row, 256 threads (3 elems/thread).
__global__ __launch_bounds__(256) void ln_kernel(
    const float* __restrict__ x, const float* __restrict__ g,
    const float* __restrict__ b, float* __restrict__ y)
{
    __shared__ float red[256];
    const int tid = threadIdx.x;
    const float* xr = x + (long long)blockIdx.x * Cch;
    float v0 = xr[tid], v1 = xr[tid + 256], v2 = xr[tid + 512];
    red[tid] = v0 + v1 + v2;
    __syncthreads();
    for (int o = 128; o > 0; o >>= 1) { if (tid < o) red[tid] += red[tid + o]; __syncthreads(); }
    const float mu = red[0] * (1.0f / Cch);
    __syncthreads();
    float d0 = v0 - mu, d1 = v1 - mu, d2 = v2 - mu;
    red[tid] = d0 * d0 + d1 * d1 + d2 * d2;
    __syncthreads();
    for (int o = 128; o > 0; o >>= 1) { if (tid < o) red[tid] += red[tid + o]; __syncthreads(); }
    const float rstd = rsqrtf(red[0] * (1.0f / Cch) + 1e-5f);
    float* yr = y + (long long)blockIdx.x * Cch;
    yr[tid]       = d0 * rstd * g[tid]       + b[tid];
    yr[tid + 256] = d1 * rstd * g[tid + 256] + b[tid + 256];
    yr[tid + 512] = d2 * rstd * g[tid + 512] + b[tid + 512];
}

// In-place row softmax over 1024 cols; one block per row, 4 elems/thread.
__global__ __launch_bounds__(256) void softmax1024_inplace(float* __restrict__ s)
{
    __shared__ float red[256];
    const int tid = threadIdx.x;
    float* p = s + (long long)blockIdx.x * 1024;
    float v[4];
    #pragma unroll
    for (int j = 0; j < 4; ++j) v[j] = p[tid + j * 256];
    float m = fmaxf(fmaxf(v[0], v[1]), fmaxf(v[2], v[3]));
    red[tid] = m;
    __syncthreads();
    for (int o = 128; o > 0; o >>= 1) { if (tid < o) red[tid] = fmaxf(red[tid], red[tid + o]); __syncthreads(); }
    m = red[0];
    __syncthreads();
    float sum = 0.0f;
    #pragma unroll
    for (int j = 0; j < 4; ++j) { v[j] = __expf(v[j] - m); sum += v[j]; }
    red[tid] = sum;
    __syncthreads();
    for (int o = 128; o > 0; o >>= 1) { if (tid < o) red[tid] += red[tid + o]; __syncthreads(); }
    const float inv = 1.0f / red[0];
    #pragma unroll
    for (int j = 0; j < 4; ++j) p[tid + j * 256] = v[j] * inv;
}

__global__ void copy2(const float* __restrict__ a, const float* __restrict__ b,
                      float* __restrict__ o1, float* __restrict__ o2, int n)
{
    int i = blockIdx.x * 256 + threadIdx.x;
    if (i < n) { o1[i] = a[i]; o2[i] = b[i]; }
}

extern "C" void kernel_launch(void* const* d_in, const int* in_sizes, int n_in,
                              void* d_out, int out_size, void* d_ws, size_t ws_size,
                              hipStream_t stream)
{
    (void)in_sizes; (void)n_in; (void)out_size; (void)ws_size;
    const float* src    = (const float*)d_in[0];
    const float* y      = (const float*)d_in[1];
    const float* shs    = (const float*)d_in[2];
    const float* pre_g  = (const float*)d_in[3];
    const float* pre_b  = (const float*)d_in[4];
    const float* qkv_w  = (const float*)d_in[5];
    const float* proj_w = (const float*)d_in[6];
    const float* proj_b = (const float*)d_in[7];
    const float* n1_g   = (const float*)d_in[8];
    const float* n1_b   = (const float*)d_in[9];
    const float* l1_w   = (const float*)d_in[10];
    const float* l1_b   = (const float*)d_in[11];
    const float* l2_w   = (const float*)d_in[12];
    const float* l2_b   = (const float*)d_in[13];

    const long long TOK = (long long)Bsz * Nsq;   // 8192
    float* ws    = (float*)d_ws;                  // fp32 workspace (~243 MB)
    float* x_ln  = ws;                            // 8192*768
    float* qkv   = x_ln  + TOK * Cch;             // 8192*2304
    float* ctx   = qkv   + TOK * 3 * Cch;         // 8192*768
    float* src2  = ctx   + TOK * Cch;             // 8192*768
    float* src_n = src2  + TOK * Cch;             // 8192*768
    float* hbuf  = src_n + TOK * Cch;             // 8192*2048

    float* out      = (float*)d_out;              // tuple: (src, shs, y, attn)
    float* out_src  = out;
    float* out_shs  = out + TOK * Cch;
    float* out_y    = out_shs + Bsz * 1000;
    float* out_attn = out_y   + Bsz * 1000;

    // 1) pre-LN
    ln_kernel<<<(int)TOK, 256, 0, stream>>>(src, pre_g, pre_b, x_ln);

    // 2) QKV: [8192,768] x [768,2304]
    gemm_wmma<EPI_NONE><<<dim3(3 * Cch / BN, TOK / BM, 1), 256, 0, stream>>>(
        x_ln, 0, 0, Cch,
        qkv_w, 0, 0, 3 * Cch, 1,
        qkv, 0, 0, 3 * Cch,
        nullptr, nullptr, 0, 0.0f, Cch);

    // 3) scores = (Q K^T) * HD^-0.5  per (b,h): [1024,64] x [64,1024]
    gemm_wmma<EPI_SCALE><<<dim3(Nsq / BN, Nsq / BM, Bsz * Hh), 256, 0, stream>>>(
        qkv, (long long)Nsq * 3 * Cch, HDd, 3 * Cch,                           // Q slice
        qkv + (long long)Hh * HDd, (long long)Nsq * 3 * Cch, HDd, 1, 3 * Cch,  // K^T (col-major view)
        out_attn, (long long)Hh * Nsq * Nsq, (long long)Nsq * Nsq, Nsq,
        nullptr, nullptr, 0, 0.125f, HDd);

    // 4) softmax rows, in place in the attn output region
    softmax1024_inplace<<<Bsz * Hh * Nsq, 256, 0, stream>>>(out_attn);

    // 5) ctx = attn @ V  per (b,h): [1024,1024] x [1024,64] -> packed [8192,768]
    gemm_wmma<EPI_NONE><<<dim3(HDd / BN, Nsq / BM, Bsz * Hh), 256, 0, stream>>>(
        out_attn, (long long)Hh * Nsq * Nsq, (long long)Nsq * Nsq, Nsq,
        qkv + (long long)2 * Hh * HDd, (long long)Nsq * 3 * Cch, HDd, 3 * Cch, 1,  // V slice
        ctx, (long long)Nsq * Cch, HDd, Cch,
        nullptr, nullptr, 0, 0.0f, Nsq);

    // 6) src2 = src + ctx @ proj_w + proj_b
    gemm_wmma<EPI_BIAS_RES><<<dim3(Cch / BN, TOK / BM, 1), 256, 0, stream>>>(
        ctx, 0, 0, Cch,
        proj_w, 0, 0, Cch, 1,
        src2, 0, 0, Cch,
        proj_b, src, Cch, 0.0f, Cch);

    // 7) post-LN
    ln_kernel<<<(int)TOK, 256, 0, stream>>>(src2, n1_g, n1_b, src_n);

    // 8) hbuf = gelu(src_n @ l1_w + l1_b)   (exact, erf-based)
    gemm_wmma<EPI_BIAS_GELU><<<dim3(FFd / BN, TOK / BM, 1), 256, 0, stream>>>(
        src_n, 0, 0, Cch,
        l1_w, 0, 0, FFd, 1,
        hbuf, 0, 0, FFd,
        l1_b, nullptr, 0, 0.0f, Cch);

    // 9) out_src = src_n + hbuf @ l2_w + l2_b
    gemm_wmma<EPI_BIAS_RES><<<dim3(Cch / BN, TOK / BM, 1), 256, 0, stream>>>(
        hbuf, 0, 0, FFd,
        l2_w, 0, 0, Cch, 1,
        out_src, 0, 0, Cch,
        l2_b, src_n, Cch, 0.0f, FFd);

    // 10) pass-through tuple members
    copy2<<<(Bsz * 1000 + 255) / 256, 256, 0, stream>>>(shs, y, out_shs, out_y, Bsz * 1000);
}
